// SelfAttention_3126736191665
// MI455X (gfx1250) — matrix operand order repaired and checked by reading, not compiled, not run on hardware
//
#include <hip/hip_runtime.h>

#define D_MODEL 1024
#define SEQ     2048
#define BATCH   2
#define NHEAD   16
#define HDIM    64
#define MTOT    (BATCH * SEQ) /* 4096 */

typedef __attribute__((ext_vector_type(16))) __bf16   v16bf;
typedef __attribute__((ext_vector_type(8)))  float    v8f;
typedef __attribute__((ext_vector_type(4)))  unsigned v4u;
typedef __attribute__((ext_vector_type(8)))  int      v8i;
typedef __attribute__((ext_vector_type(4)))  int      v4i;

// 16-lane fragment assembled from two 16-byte chunks -> ds_load_b128 x2
union Frag {
  v16bf v;
  uint4 q[2];
};

// ---------- TDM availability (device pass only) ----------
#if defined(__gfx1250__) && defined(__has_builtin)
#if __has_builtin(__builtin_amdgcn_tensor_load_to_lds) && \
    __has_builtin(__builtin_amdgcn_s_wait_tensorcnt)
#define HAVE_TDM 1
#endif
#endif
#ifndef HAVE_TDM
#define HAVE_TDM 0
#endif

#if HAVE_TDM
// Issue one 2D TDM tile load: global (row-major, elem=2B) -> LDS (+row pad).
// padIntervalCode: pad after 2^(c+1) DWORDs; padAmountCode: (pad DWORDs)-1.
static __device__ __forceinline__ void tdm_load_2d(
    unsigned ldsByteAddr, const void* gaddr, unsigned tensorDim0,
    unsigned tensorDim1, unsigned tileDim0, unsigned tileDim1,
    unsigned stride0, unsigned padIntervalCode, unsigned padAmountCode) {
  unsigned long long ga = (unsigned long long)(uintptr_t)gaddr;
  v4u g0;
  g0[0] = 1u;  // count=1, user mode, no gather
  g0[1] = ldsByteAddr;
  g0[2] = (unsigned)(ga & 0xFFFFFFFFu);
  g0[3] = ((unsigned)((ga >> 32) & 0x01FFFFFFu)) | (2u << 30);  // type=2
  v8i g1;
  g1[0] = (int)((1u << 16) /*data_size=2B*/ | (1u << 20) /*pad_enable*/ |
                (padIntervalCode << 22) | (padAmountCode << 25));
  g1[1] = (int)((tensorDim0 & 0xFFFFu) << 16);
  g1[2] = (int)((tensorDim0 >> 16) | ((tensorDim1 & 0xFFFFu) << 16));
  g1[3] = (int)((tensorDim1 >> 16) | (tileDim0 << 16));
  g1[4] = (int)tileDim1;  // tile_dim1; tile_dim2=0
  g1[5] = (int)stride0;   // tensor_dim0_stride (low 32)
  g1[6] = 0;
  g1[7] = 0;
  v4i gz;
  gz[0] = gz[1] = gz[2] = gz[3] = 0;
#if __clang_major__ >= 23
  v8i gz8;
  gz8[0] = gz8[1] = gz8[2] = gz8[3] = gz8[4] = gz8[5] = gz8[6] = gz8[7] = 0;
  __builtin_amdgcn_tensor_load_to_lds(g0, g1, gz, gz, gz8, 0);
#else
  __builtin_amdgcn_tensor_load_to_lds(g0, g1, gz, gz, 0);
#endif
}
#endif  // HAVE_TDM

// ---------- helpers ----------
static __device__ __forceinline__ unsigned short f2bf(float f) {
  unsigned u = __float_as_uint(f);
  u += 0x7FFFu + ((u >> 16) & 1u);  // round-to-nearest-even
  return (unsigned short)(u >> 16);
}

// ---------- kernel 1: f32 -> bf16 (no transpose, for x) ----------
__global__ void cvt_bf16_kernel(const float* __restrict__ src,
                                unsigned short* __restrict__ dst, int n) {
  int i = blockIdx.x * blockDim.x + threadIdx.x;
  int stride = gridDim.x * blockDim.x;
  for (; i < n; i += stride) dst[i] = f2bf(src[i]);
}

// ---------- kernel 1b: f32 W[k][n] -> bf16 Wt[n][k] (tiled transpose) ----
__global__ void __launch_bounds__(256)
cvtT_bf16_kernel(const float* __restrict__ in,
                 unsigned short* __restrict__ out) {
  __shared__ unsigned short t[64][72];
  const int kb = blockIdx.y * 64, nb = blockIdx.x * 64;
#pragma unroll
  for (int it = 0; it < 16; ++it) {
    int idx = threadIdx.x + it * 256;
    int k = idx >> 6, n = idx & 63;
    t[k][n] = f2bf(in[(size_t)(kb + k) * D_MODEL + nb + n]);
  }
  __syncthreads();
#pragma unroll
  for (int it = 0; it < 16; ++it) {
    int idx = threadIdx.x + it * 256;
    int n = idx >> 6, k = idx & 63;
    out[(size_t)(nb + n) * D_MODEL + kb + k] = t[k][n];
  }
}

// ---------- kernel 1c: bf16 V [B*S][D] -> Vt [BH][HDIM][SEQ] ----------
__global__ void __launch_bounds__(256)
transposeV_kernel(const unsigned short* __restrict__ in,
                  unsigned short* __restrict__ out) {
  __shared__ unsigned short t[64][72];
  const int sBlk = blockIdx.x * 64;
  const int bh = blockIdx.y;
  const int b = bh >> 4, h = bh & 15;
#pragma unroll
  for (int it = 0; it < 16; ++it) {
    int idx = threadIdx.x + it * 256;
    int s = idx >> 6, d = idx & 63;
    t[s][d] = in[((size_t)b * SEQ + sBlk + s) * D_MODEL + h * HDIM + d];
  }
  __syncthreads();
#pragma unroll
  for (int it = 0; it < 16; ++it) {
    int idx = threadIdx.x + it * 256;
    int d = idx >> 6, s = idx & 63;
    out[((size_t)bh * HDIM + d) * SEQ + sBlk + s] = t[s][d];
  }
}

// ---------- kernel 2: bf16 GEMM  C = A * Bt^T + bias ----------
// A: [M][Kd] bf16, Bt: [N][Kd] bf16 (pre-transposed weights).
// Block tile 128x128, K-step 32; 8 waves = 4(M) x 2(N), 32x64 per wave.
#define BM  128
#define BN  128
#define BKs 32
#define LDA 40  // u16 row stride (80B = 64B data + 16B pad, 16B-aligned)

__global__ void __launch_bounds__(256)
gemm_bf16_kernel(const unsigned short* __restrict__ A,
                 const unsigned short* __restrict__ Bt,
                 const float* __restrict__ bias,
                 unsigned short* __restrict__ Cb,  // bf16 out (or null)
                 float* __restrict__ Cf,           // f32 out (or null)
                 int M, int N, int Kd) {
  __shared__ unsigned short As[BM][LDA];  // [m][k]
  __shared__ unsigned short Bs[BN][LDA];  // [n][k]  (K contiguous)

  const int tid    = threadIdx.x;
  const int lane   = tid & 31;
  const int lane16 = lane & 15;
  const int laneHi = lane >> 4;
  const int wave   = tid >> 5;
  const int waveM  = wave & 3;
  const int waveN  = wave >> 2;
  const int rowBase = blockIdx.y * BM;
  const int colBase = blockIdx.x * BN;

  v8f acc[2][4];
#pragma unroll
  for (int mi = 0; mi < 2; ++mi)
#pragma unroll
    for (int ni = 0; ni < 4; ++ni) acc[mi][ni] = {};

  for (int k0 = 0; k0 < Kd; k0 += BKs) {
#if HAVE_TDM
    // Tensor Data Mover: one wave DMAs both 128x32 tiles into padded LDS.
    // 64B rows = 16 DWORDs -> pad_interval code 3; pad 16B = 4 DWORDs -> 3.
    if (wave == 0) {
      tdm_load_2d((unsigned)(uintptr_t)&As[0][0],
                  &A[(size_t)rowBase * Kd + k0], (unsigned)Kd, (unsigned)M,
                  BKs, BM, (unsigned)Kd, 3u, 3u);
      tdm_load_2d((unsigned)(uintptr_t)&Bs[0][0],
                  &Bt[(size_t)colBase * Kd + k0], (unsigned)Kd, (unsigned)N,
                  BKs, BN, (unsigned)Kd, 3u, 3u);
      __builtin_amdgcn_s_wait_tensorcnt(0);
    }
#else
    if (k0 + BKs < Kd) {
      __builtin_prefetch(&A[(size_t)rowBase * Kd + k0 + BKs], 0, 3);
      __builtin_prefetch(&Bt[(size_t)colBase * Kd + k0 + BKs], 0, 3);
    }
#pragma unroll
    for (int it = 0; it < 2; ++it) {
      int idx = tid + it * 256;
      int r = idx >> 2, cq = idx & 3;  // 4 uint4 per row
      *reinterpret_cast<uint4*>(&As[r][cq * 8]) =
          *reinterpret_cast<const uint4*>(
              &A[(size_t)(rowBase + r) * Kd + k0 + cq * 8]);
      *reinterpret_cast<uint4*>(&Bs[r][cq * 8]) =
          *reinterpret_cast<const uint4*>(
              &Bt[(size_t)(colBase + r) * Kd + k0 + cq * 8]);
    }
#endif
    __syncthreads();  // tiles ready

    Frag afrag[2], bfrag[4];
#pragma unroll
    for (int mi = 0; mi < 2; ++mi) {
      int m = waveM * 32 + mi * 16 + lane16;
      afrag[mi].q[0] = *reinterpret_cast<const uint4*>(&As[m][laneHi * 8]);
      afrag[mi].q[1] = *reinterpret_cast<const uint4*>(&As[m][laneHi * 8 + 16]);
    }
#pragma unroll
    for (int ni = 0; ni < 4; ++ni) {
      int n = waveN * 64 + ni * 16 + lane16;
      bfrag[ni].q[0] = *reinterpret_cast<const uint4*>(&Bs[n][laneHi * 8]);
      bfrag[ni].q[1] = *reinterpret_cast<const uint4*>(&Bs[n][laneHi * 8 + 16]);
    }
#pragma unroll
    for (int mi = 0; mi < 2; ++mi)
#pragma unroll
      for (int ni = 0; ni < 4; ++ni)
        acc[mi][ni] = __builtin_amdgcn_wmma_f32_16x16x32_bf16(
            false, afrag[mi].v, false, bfrag[ni].v, (short)0, acc[mi][ni],
            false, false);
    __syncthreads();  // safe to overwrite tiles
  }

  // epilogue: element i -> row (i + laneHi*8), lane16 -> col
#pragma unroll
  for (int mi = 0; mi < 2; ++mi)
#pragma unroll
    for (int ni = 0; ni < 4; ++ni) {
      int col = colBase + waveN * 64 + ni * 16 + lane16;
      float bv = bias ? bias[col] : 0.0f;
#pragma unroll
      for (int i = 0; i < 8; ++i) {
        int row = rowBase + waveM * 32 + mi * 16 + laneHi * 8 + i;
        float v = acc[mi][ni][i] + bv;
        if (Cf) Cf[(size_t)row * N + col] = v;
        else    Cb[(size_t)row * N + col] = f2bf(v);
      }
    }
}

// ---------- kernel 3: flash attention ----------
// grid: (SEQ/64, BATCH*NHEAD); block: 128 (4 waves, 16 q-rows each)
#define KT  64
#define LDK 72  // u16 row stride (144B = 128B data + 16B pad, 16B-aligned)

__global__ void __launch_bounds__(128)
attn_kernel(const unsigned short* __restrict__ Qg,   // [B*S][D]
            const unsigned short* __restrict__ Kg,   // [B*S][D]
            const unsigned short* __restrict__ Vtg,  // [BH][HDIM][SEQ]
            unsigned short* __restrict__ Og) {       // [B*S][D]
  __shared__ unsigned short Kt[KT][LDK];  // [key][d]  (d contiguous)
  __shared__ unsigned short Vt[KT][LDK];  // [d][key]  (key contiguous)
  __shared__ unsigned short Ps[4][16][LDK];

  const int tid    = threadIdx.x;
  const int lane   = tid & 31;
  const int lane16 = lane & 15;
  const int laneHi = lane >> 4;
  const int wave   = tid >> 5;
  const int qTile  = blockIdx.x;
  const int bh     = blockIdx.y;
  const int b      = bh >> 4;
  const int h      = bh & 15;
  const int colOff = h * HDIM;
  const size_t rowBaseQ  = (size_t)b * SEQ + qTile * 64 + wave * 16;
  const size_t rowBaseBH = (size_t)b * SEQ;
  const size_t vBase     = (size_t)bh * HDIM * SEQ;
  const float scale = 0.125f;  // 1/sqrt(64)

  // Q fragments: two k-steps, each = two global b128 loads
  Frag qfrag[2];
#pragma unroll
  for (int ks = 0; ks < 2; ++ks) {
    const unsigned short* qp =
        &Qg[(rowBaseQ + lane16) * D_MODEL + colOff + ks * 32 + laneHi * 8];
    qfrag[ks].q[0] = *reinterpret_cast<const uint4*>(qp);
    qfrag[ks].q[1] = *reinterpret_cast<const uint4*>(qp + 16);
  }

  v8f Oacc[4];
#pragma unroll
  for (int nt = 0; nt < 4; ++nt) Oacc[nt] = {};
  float mrun[8], lrun[8];
#pragma unroll
  for (int i = 0; i < 8; ++i) { mrun[i] = -3.0e38f; lrun[i] = 0.0f; }

  for (int kb = 0; kb < SEQ / KT; ++kb) {
    __syncthreads();  // previous tiles fully consumed
#if HAVE_TDM
    // TDM: 64x64 tiles, 128B rows = 32 DWORDs -> interval code 4; pad 16B.
    if (wave == 0) {
      tdm_load_2d((unsigned)(uintptr_t)&Kt[0][0],
                  &Kg[(rowBaseBH + (size_t)kb * KT) * D_MODEL + colOff],
                  D_MODEL, MTOT, KT, KT, D_MODEL, 4u, 3u);
      tdm_load_2d((unsigned)(uintptr_t)&Vt[0][0],
                  &Vtg[vBase + (size_t)kb * KT],
                  SEQ, BATCH * NHEAD * HDIM, KT, KT, SEQ, 4u, 3u);
      __builtin_amdgcn_s_wait_tensorcnt(0);
    }
#else
#pragma unroll
    for (int it = 0; it < 4; ++it) {
      int idx = tid + it * 128;
      int r = idx >> 3, cq = idx & 7;  // 8 uint4 per row
      *reinterpret_cast<uint4*>(&Kt[r][cq * 8]) =
          *reinterpret_cast<const uint4*>(
              &Kg[(rowBaseBH + (size_t)kb * KT + r) * D_MODEL + colOff +
                  cq * 8]);
      *reinterpret_cast<uint4*>(&Vt[r][cq * 8]) =
          *reinterpret_cast<const uint4*>(
              &Vtg[vBase + (size_t)r * SEQ + (size_t)kb * KT + cq * 8]);
    }
#endif
    __syncthreads();

    // S = Q K^T ; B-frag: n=key(lane16), k=d contiguous -> b128 x2
    v8f accS[4];
#pragma unroll
    for (int nt = 0; nt < 4; ++nt) {
      accS[nt] = {};
#pragma unroll
      for (int ks = 0; ks < 2; ++ks) {
        Frag bf;
        const unsigned short* kp = &Kt[nt * 16 + lane16][ks * 32 + laneHi * 8];
        bf.q[0] = *reinterpret_cast<const uint4*>(kp);
        bf.q[1] = *reinterpret_cast<const uint4*>(kp + 16);
        accS[nt] = __builtin_amdgcn_wmma_f32_16x16x32_bf16(
            false, qfrag[ks].v, false, bf.v, (short)0, accS[nt], false, false);
      }
    }

    // online softmax (row = acc element, cols striped over 16 lanes)
    float pv[4][8];
#pragma unroll
    for (int i = 0; i < 8; ++i) {
      float mb = -3.0e38f;
#pragma unroll
      for (int nt = 0; nt < 4; ++nt) {
        float s = accS[nt][i] * scale;
        pv[nt][i] = s;
        mb = fmaxf(mb, s);
      }
#pragma unroll
      for (int off = 1; off < 16; off <<= 1)
        mb = fmaxf(mb, __shfl_xor(mb, off, 32));
      float mnew = fmaxf(mrun[i], mb);
      float corr = __expf(mrun[i] - mnew);
      mrun[i] = mnew;
      float rs = 0.0f;
#pragma unroll
      for (int nt = 0; nt < 4; ++nt) {
        float p = __expf(pv[nt][i] - mnew);
        pv[nt][i] = p;
        rs += p;
      }
#pragma unroll
      for (int off = 1; off < 16; off <<= 1)
        rs += __shfl_xor(rs, off, 32);
      lrun[i] = lrun[i] * corr + rs;
#pragma unroll
      for (int nt = 0; nt < 4; ++nt) Oacc[nt][i] = Oacc[nt][i] * corr;
    }

    // stage P (C-layout -> row-major LDS) for A-fragment reload
#pragma unroll
    for (int nt = 0; nt < 4; ++nt)
#pragma unroll
      for (int i = 0; i < 8; ++i)
        Ps[wave][laneHi * 8 + i][nt * 16 + lane16] = f2bf(pv[nt][i]);
    __syncthreads();

    // O += P * V ; P-frag: two 8-elem runs -> b128 x2; V-frag: key contiguous
#pragma unroll
    for (int ks = 0; ks < 2; ++ks) {
      Frag pf;
      const unsigned short* pp = &Ps[wave][lane16][ks * 32 + laneHi * 8];
      pf.q[0] = *reinterpret_cast<const uint4*>(pp);
      pf.q[1] = *reinterpret_cast<const uint4*>(pp + 16);
#pragma unroll
      for (int nt = 0; nt < 4; ++nt) {
        Frag vf;
        const unsigned short* vp = &Vt[nt * 16 + lane16][ks * 32 + laneHi * 8];
        vf.q[0] = *reinterpret_cast<const uint4*>(vp);
        vf.q[1] = *reinterpret_cast<const uint4*>(vp + 16);
        Oacc[nt] = __builtin_amdgcn_wmma_f32_16x16x32_bf16(
            false, pf.v, false, vf.v, (short)0, Oacc[nt], false, false);
      }
    }
  }

  // finalize: O /= l, write bf16
#pragma unroll
  for (int i = 0; i < 8; ++i) {
    float inv = 1.0f / lrun[i];
    size_t row = rowBaseQ + laneHi * 8 + i;
#pragma unroll
    for (int nt = 0; nt < 4; ++nt)
      Og[row * D_MODEL + colOff + nt * 16 + lane16] = f2bf(Oacc[nt][i] * inv);
  }
}

// ---------- launch ----------
extern "C" void kernel_launch(void* const* d_in, const int* in_sizes, int n_in,
                              void* d_out, int out_size, void* d_ws,
                              size_t ws_size, hipStream_t stream) {
  (void)in_sizes; (void)n_in; (void)out_size; (void)ws_size;
  const float* x  = (const float*)d_in[0];
  const float* Wq = (const float*)d_in[1];
  const float* bq = (const float*)d_in[2];
  const float* Wk = (const float*)d_in[3];
  const float* bk = (const float*)d_in[4];
  const float* Wv = (const float*)d_in[5];
  const float* bv = (const float*)d_in[6];
  const float* Wo = (const float*)d_in[7];
  const float* bo = (const float*)d_in[8];
  float* out = (float*)d_out;

  const size_t XE = (size_t)MTOT * D_MODEL;     // 4,194,304
  const size_t WE = (size_t)D_MODEL * D_MODEL;  // 1,048,576
  unsigned short* ws  = (unsigned short*)d_ws;
  unsigned short* Xb  = ws;       // x bf16
  unsigned short* Wqt = Xb + XE;  // W^T bf16 [N][K]
  unsigned short* Wkt = Wqt + WE;
  unsigned short* Wvt = Wkt + WE;
  unsigned short* Wot = Wvt + WE;
  unsigned short* Qb  = Wot + WE;
  unsigned short* Kb  = Qb + XE;
  unsigned short* Vb  = Kb + XE;
  unsigned short* Ob  = Vb + XE;
  unsigned short* Vtg = Ob + XE;  // V^T per (b,h): [32][64][2048]
  // total ~56 MB of d_ws

  cvt_bf16_kernel<<<1024, 256, 0, stream>>>(x, Xb, (int)XE);
  dim3 gT(D_MODEL / 64, D_MODEL / 64);  // (16,16)
  cvtT_bf16_kernel<<<gT, 256, 0, stream>>>(Wq, Wqt);
  cvtT_bf16_kernel<<<gT, 256, 0, stream>>>(Wk, Wkt);
  cvtT_bf16_kernel<<<gT, 256, 0, stream>>>(Wv, Wvt);
  cvtT_bf16_kernel<<<gT, 256, 0, stream>>>(Wo, Wot);

  dim3 gGemm(D_MODEL / BN, MTOT / BM);  // (8, 32)
  gemm_bf16_kernel<<<gGemm, 256, 0, stream>>>(Xb, Wqt, bq, Qb, nullptr,
                                              MTOT, D_MODEL, D_MODEL);
  gemm_bf16_kernel<<<gGemm, 256, 0, stream>>>(Xb, Wkt, bk, Kb, nullptr,
                                              MTOT, D_MODEL, D_MODEL);
  gemm_bf16_kernel<<<gGemm, 256, 0, stream>>>(Xb, Wvt, bv, Vb, nullptr,
                                              MTOT, D_MODEL, D_MODEL);

  dim3 gTV(SEQ / 64, BATCH * NHEAD);  // (32, 32)
  transposeV_kernel<<<gTV, 256, 0, stream>>>(Vb, Vtg);

  dim3 gAttn(SEQ / 64, BATCH * NHEAD);  // (32, 32)
  attn_kernel<<<gAttn, 128, 0, stream>>>(Qb, Kb, Vtg, Ob);

  gemm_bf16_kernel<<<gGemm, 256, 0, stream>>>(Ob, Wot, bo, nullptr, out,
                                              MTOT, D_MODEL, D_MODEL);
}
